// MambaBlock2_17523466568294
// MI455X (gfx1250) — compile-verified, matrix operand order checked by reading, not run on hardware
//
#include <hip/hip_runtime.h>

#define T_LEN   2048
#define B_SZ    4
#define D_MODEL 1024
#define D_INNER 2048
#define D_STATE 16
#define D_CONV  4
#define DT_RANK 64
#define M_ROWS  (T_LEN * B_SZ)           // 8192
#define XDBL_N  (DT_RANK + 2 * D_STATE)  // 96

typedef __bf16 bf16;
typedef bf16  v8bf  __attribute__((ext_vector_type(8)));
typedef bf16  v16bf __attribute__((ext_vector_type(16)));
typedef float v8f   __attribute__((ext_vector_type(8)));
typedef int   i32x4 __attribute__((ext_vector_type(4)));

#if defined(__AMDGCN__) && __has_builtin(__builtin_amdgcn_global_load_async_to_lds_b128)
#define HAVE_ASYNC 1
#else
#define HAVE_ASYNC 0
#endif

__device__ __forceinline__ float silu_f(float x) {
  return x / (1.0f + __expf(-x));
}
__device__ __forceinline__ float softplus_f(float x) {
  return (x > 20.0f) ? x : log1pf(__expf(x));
}

// ---- WMMA fragment helpers -------------------------------------------------
// A-operand register layout (ISA 7.12.2): lane = matrix row (lane&15); lanes
// 0-15 hold K = k0+0..7 (VGPR0-3) and k0+16..23 (VGPR4-7); lanes 16-31 hold
// k0+8..15 and k0+24..31. B is loaded column-major (rows of W^T) with the
// symmetric layout.
__device__ __forceinline__ v16bf frag_from(const bf16* p /* row base + half*8 */) {
  const v8bf* q = (const v8bf*)p;
  v8bf lo = q[0];   // k = half*8 .. +7
  v8bf hi = q[2];   // k = 16 + half*8 .. +7
  return __builtin_shufflevector(lo, hi,
      0, 1, 2, 3, 4, 5, 6, 7, 8, 9, 10, 11, 12, 13, 14, 15);
}

__device__ __forceinline__ v16bf load_frag_g(const bf16* __restrict__ base,
                                             int row, int K, int k0, int half) {
  return frag_from(base + (size_t)row * K + k0 + half * 8);
}

// LDS tile rows are 32 bf16 (64 B) each.
__device__ __forceinline__ v16bf load_frag_l(const bf16* lds_base, int row, int half) {
  return frag_from(lds_base + row * 32 + half * 8);
}

// ---- global -> LDS staging (async engine if available) ---------------------
__device__ __forceinline__ void stage_chunk(const bf16* __restrict__ g, bf16* l) {
#if HAVE_ASYNC
  // (global src, lds dst, imm offset, imm cpol) -- params are i32x4*.
  __builtin_amdgcn_global_load_async_to_lds_b128((i32x4*)g, (i32x4*)l, 0, 0);
#else
  *(v8bf*)l = *(const v8bf*)g;     // global_load_b128 + ds_store_b128
#endif
}

__device__ __forceinline__ void wait_async0() {
#if HAVE_ASYNC
#if __has_builtin(__builtin_amdgcn_s_wait_asynccnt)
  __builtin_amdgcn_s_wait_asynccnt(0);
#else
  asm volatile("s_wait_asynccnt 0x0" ::: "memory");
#endif
#endif
}

// ---- epilogue --------------------------------------------------------------
template <int MODE>
__device__ __forceinline__ void store_tile(float* __restrict__ C0,
                                           const float* __restrict__ bias,
                                           int N, int mbase, int ncol, int half,
                                           v8f acc) {
  // C/D layout: lane = N-column (lane&15), VGPR v holds M = v + 8*(lane>>4).
#pragma unroll
  for (int v = 0; v < 8; v++) {
    int mr = mbase + v + half * 8;
    float val = acc[v];
    if (MODE == 0) {
      C0[(size_t)mr * N + ncol] = val;
    } else if (MODE == 1) {
      C0[(size_t)mr * N + ncol] = softplus_f(val + bias[ncol]);
    } else {  // MODE 2: scatter rows (b*T + t) -> out[t, b, ncol]
      int bb = mr >> 11;            // mr / T_LEN
      int tt = mr & (T_LEN - 1);
      C0[((size_t)tt * B_SZ + bb) * D_MODEL + ncol] = val;
    }
  }
}

// ---- LDS-staged, double-buffered GEMM --------------------------------------
// Workgroup = 8 waves computing a 128x64 tile (4x2 grid of 32x32 wave tiles).
// Per 32-wide K stage: A tile 128x32 bf16 (8 KB), B tile 64x32 bf16 (4 KB),
// staged via 3 async b128 loads per thread, double-buffered (24 KB LDS).
// Requires M % 128 == 0, N % 64 == 0, K % 32 == 0.
template <int MODE>
__global__ __launch_bounds__(256) void gemm_bf16_lds(
    const bf16* __restrict__ A, const bf16* __restrict__ WT, int M, int N,
    int K, float* __restrict__ C0, const float* __restrict__ bias) {
  __shared__ bf16 smA[2][128 * 32];
  __shared__ bf16 smB[2][64 * 32];

  const int tid = threadIdx.x;
  const int lane = tid & 31;
  const int wave = tid >> 5;
  const int ntn = N >> 6;
  const int tm = blockIdx.x / ntn;
  const int tn = blockIdx.x - tm * ntn;

  const bf16* gA = A + (size_t)tm * 128 * K;
  const bf16* gB = WT + (size_t)tn * 64 * K;

  // staging assignment: A has 512 b128 chunks (row = c>>2, seg = c&3),
  // B has 256 chunks (row = tid>>2, seg = tid&3).
  const int ac0 = tid * 2;
  const int brow = tid >> 2;
  const int bseg = tid & 3;

  auto stage = [&](int buf, int k0) {
#pragma unroll
    for (int j = 0; j < 2; j++) {
      int c = ac0 + j;
      int row = c >> 2, seg = c & 3;
      stage_chunk(gA + (size_t)row * K + k0 + seg * 8,
                  &smA[buf][row * 32 + seg * 8]);
    }
    stage_chunk(gB + (size_t)brow * K + k0 + bseg * 8,
                &smB[buf][brow * 32 + bseg * 8]);
  };

  const int lrow = lane & 15;
  const int half = lane >> 4;
  const int wm = wave & 3;        // 0..3 -> M sub-tile
  const int wn = wave >> 2;       // 0..1 -> N sub-tile
  const int ra0 = wm * 32 + lrow, ra1 = ra0 + 16;
  const int rb0 = wn * 32 + lrow, rb1 = rb0 + 16;

  v8f acc00 = {}, acc01 = {}, acc10 = {}, acc11 = {};

  stage(0, 0);                     // prologue: stage K-step 0
  const int nk = K >> 5;
  for (int i = 0; i < nk; i++) {
    wait_async0();                 // own stage-i writes landed in LDS
    __syncthreads();               // everyone's stage-i tile visible
    if (i + 1 < nk) stage((i + 1) & 1, (i + 1) * 32);  // overlap next stage

    const bf16* bufA = &smA[i & 1][0];
    const bf16* bufB = &smB[i & 1][0];
    v16bf a0 = load_frag_l(bufA, ra0, half);
    v16bf a1 = load_frag_l(bufA, ra1, half);
    v16bf b0 = load_frag_l(bufB, rb0, half);
    v16bf b1 = load_frag_l(bufB, rb1, half);

    acc00 = __builtin_amdgcn_wmma_f32_16x16x32_bf16(false, a0, false, b0,
                                                    (short)0, acc00, false, false);
    acc01 = __builtin_amdgcn_wmma_f32_16x16x32_bf16(false, a0, false, b1,
                                                    (short)0, acc01, false, false);
    acc10 = __builtin_amdgcn_wmma_f32_16x16x32_bf16(false, a1, false, b0,
                                                    (short)0, acc10, false, false);
    acc11 = __builtin_amdgcn_wmma_f32_16x16x32_bf16(false, a1, false, b1,
                                                    (short)0, acc11, false, false);
  }

  const int mb0 = tm * 128 + wm * 32;
  const int nc0 = tn * 64 + wn * 32 + lrow;
  store_tile<MODE>(C0, bias, N, mb0,      nc0,      half, acc00);
  store_tile<MODE>(C0, bias, N, mb0,      nc0 + 16, half, acc01);
  store_tile<MODE>(C0, bias, N, mb0 + 16, nc0,      half, acc10);
  store_tile<MODE>(C0, bias, N, mb0 + 16, nc0 + 16, half, acc11);
}

// ---- direct-load GEMM (used when N % 64 != 0, i.e. x_proj N=96) ------------
template <int MODE>
__global__ __launch_bounds__(256) void gemm_bf16_direct(
    const bf16* __restrict__ A, const bf16* __restrict__ WT, int M, int N,
    int K, float* __restrict__ C0, const float* __restrict__ bias) {
  const int lane = threadIdx.x & 31;
  const int wave = threadIdx.x >> 5;
  const int ntn = N >> 5;
  const int tile = blockIdx.x * 8 + wave;
  if (tile >= (M >> 5) * ntn) return;
  const int tm = tile / ntn;
  const int tn = tile - tm * ntn;

  const int lrow = lane & 15;
  const int half = lane >> 4;
  const int ar0 = tm * 32 + lrow, ar1 = ar0 + 16;
  const int br0 = tn * 32 + lrow, br1 = br0 + 16;

  v8f acc00 = {}, acc01 = {}, acc10 = {}, acc11 = {};
  for (int k0 = 0; k0 < K; k0 += 32) {
    v16bf a0 = load_frag_g(A, ar0, K, k0, half);
    v16bf a1 = load_frag_g(A, ar1, K, k0, half);
    v16bf b0 = load_frag_g(WT, br0, K, k0, half);
    v16bf b1 = load_frag_g(WT, br1, K, k0, half);
    if (k0 + 64 < K) {
      __builtin_prefetch(A + (size_t)ar0 * K + k0 + 64, 0, 1);
      __builtin_prefetch(A + (size_t)ar1 * K + k0 + 64, 0, 1);
      __builtin_prefetch(WT + (size_t)br0 * K + k0 + 64, 0, 1);
      __builtin_prefetch(WT + (size_t)br1 * K + k0 + 64, 0, 1);
    }
    acc00 = __builtin_amdgcn_wmma_f32_16x16x32_bf16(false, a0, false, b0,
                                                    (short)0, acc00, false, false);
    acc01 = __builtin_amdgcn_wmma_f32_16x16x32_bf16(false, a0, false, b1,
                                                    (short)0, acc01, false, false);
    acc10 = __builtin_amdgcn_wmma_f32_16x16x32_bf16(false, a1, false, b0,
                                                    (short)0, acc10, false, false);
    acc11 = __builtin_amdgcn_wmma_f32_16x16x32_bf16(false, a1, false, b1,
                                                    (short)0, acc11, false, false);
  }

  store_tile<MODE>(C0, bias, N, tm * 32,      tn * 32 + lrow,      half, acc00);
  store_tile<MODE>(C0, bias, N, tm * 32,      tn * 32 + 16 + lrow, half, acc01);
  store_tile<MODE>(C0, bias, N, tm * 32 + 16, tn * 32 + lrow,      half, acc10);
  store_tile<MODE>(C0, bias, N, tm * 32 + 16, tn * 32 + 16 + lrow, half, acc11);
}

// ---- elementwise / conversion kernels --------------------------------------
__global__ __launch_bounds__(256) void k_transpose_bf16(
    const float* __restrict__ W, bf16* __restrict__ WT, int K, int N) {
  int idx = blockIdx.x * 256 + threadIdx.x;
  if (idx >= K * N) return;
  int k = idx / N;
  int n = idx - k * N;
  WT[(size_t)n * K + k] = (bf16)W[idx];
}

__global__ __launch_bounds__(256) void k_gather_x_bf16(
    const float* __restrict__ x, bf16* __restrict__ xbf) {
  int idx = blockIdx.x * 256 + threadIdx.x;
  if (idx >= T_LEN * B_SZ * D_MODEL) return;
  int c = idx & (D_MODEL - 1);
  int tb = idx >> 10;
  int b = tb & (B_SZ - 1);
  int t = tb >> 2;
  xbf[((size_t)b * T_LEN + t) * D_MODEL + c] = (bf16)x[idx];
}

__global__ __launch_bounds__(256) void k_conv_silu(
    const float* __restrict__ y1, const float* __restrict__ cw,
    const float* __restrict__ cb, float* __restrict__ u,
    bf16* __restrict__ ubf) {
  int idx = blockIdx.x * 256 + threadIdx.x;
  if (idx >= M_ROWS * D_INNER) return;
  int d = idx & (D_INNER - 1);
  int r = idx >> 11;
  int t = r & (T_LEN - 1);
  float acc = cb[d];
#pragma unroll
  for (int j = 0; j < D_CONV; j++) {
    int tt = t - (D_CONV - 1) + j;
    if (tt >= 0)
      acc += cw[d * D_CONV + j] *
             y1[(size_t)(r - (D_CONV - 1) + j) * (2 * D_INNER) + d];
  }
  float s = silu_f(acc);
  u[idx] = s;
  ubf[idx] = (bf16)s;
}

__global__ __launch_bounds__(256) void k_slice_dta(
    const float* __restrict__ xdbl, bf16* __restrict__ dtabf) {
  int idx = blockIdx.x * 256 + threadIdx.x;
  if (idx >= M_ROWS * DT_RANK) return;
  int c = idx & (DT_RANK - 1);
  int r = idx >> 6;
  dtabf[idx] = (bf16)xdbl[(size_t)r * XDBL_N + c];
}

// Selective scan: one thread per (b, d) channel, 16-state recurrence in
// registers, fused skip (u*D) and gate (silu(res)); lanes in a wave share b so
// B/C loads broadcast.
__global__ __launch_bounds__(256) void k_scan(
    const float* __restrict__ delta, const float* __restrict__ u,
    const float* __restrict__ xdbl, const float* __restrict__ y1,
    const float* __restrict__ A_log, const float* __restrict__ Dv,
    bf16* __restrict__ ybf) {
  int ch = blockIdx.x * 256 + threadIdx.x;
  if (ch >= B_SZ * D_INNER) return;
  int b = ch >> 11;
  int d = ch & (D_INNER - 1);

  float Arow[D_STATE];
#pragma unroll
  for (int n = 0; n < D_STATE; n++) Arow[n] = -__expf(A_log[d * D_STATE + n]);
  const float Dd = Dv[d];
  float h[D_STATE];
#pragma unroll
  for (int n = 0; n < D_STATE; n++) h[n] = 0.0f;

  const size_t rb = (size_t)b * T_LEN;
  for (int t = 0; t < T_LEN; t++) {
    const size_t r = rb + t;
    float dlt = delta[r * D_INNER + d];
    float uu = u[r * D_INNER + d];
    float du = dlt * uu;
    const float4* Bv = (const float4*)(xdbl + r * XDBL_N + DT_RANK);
    const float4* Cv = (const float4*)(xdbl + r * XDBL_N + DT_RANK + D_STATE);
    float y = 0.0f;
#pragma unroll
    for (int q = 0; q < 4; q++) {
      float4 Bq = Bv[q];
      float4 Cq = Cv[q];
      float dA;
      dA = __expf(dlt * Arow[4 * q + 0]); h[4 * q + 0] = dA * h[4 * q + 0] + du * Bq.x; y += h[4 * q + 0] * Cq.x;
      dA = __expf(dlt * Arow[4 * q + 1]); h[4 * q + 1] = dA * h[4 * q + 1] + du * Bq.y; y += h[4 * q + 1] * Cq.y;
      dA = __expf(dlt * Arow[4 * q + 2]); h[4 * q + 2] = dA * h[4 * q + 2] + du * Bq.z; y += h[4 * q + 2] * Cq.z;
      dA = __expf(dlt * Arow[4 * q + 3]); h[4 * q + 3] = dA * h[4 * q + 3] + du * Bq.w; y += h[4 * q + 3] * Cq.w;
    }
    float resv = y1[r * (2 * D_INNER) + D_INNER + d];
    float outv = (y + uu * Dd) * silu_f(resv);
    ybf[r * D_INNER + d] = (bf16)outv;
  }
}

// ---- host-side orchestration ----------------------------------------------
extern "C" void kernel_launch(void* const* d_in, const int* in_sizes, int n_in,
                              void* d_out, int out_size, void* d_ws,
                              size_t ws_size, hipStream_t stream) {
  const float* x      = (const float*)d_in[0];
  const float* W_in   = (const float*)d_in[1];
  const float* conv_w = (const float*)d_in[2];
  const float* conv_b = (const float*)d_in[3];
  const float* W_x    = (const float*)d_in[4];
  const float* W_dt   = (const float*)d_in[5];
  const float* b_dt   = (const float*)d_in[6];
  const float* A_log  = (const float*)d_in[7];
  const float* Dvec   = (const float*)d_in[8];
  const float* W_out  = (const float*)d_in[9];
  float* out = (float*)d_out;

  char* base = (char*)d_ws;
  size_t off = 0;
  auto carve = [&](size_t bytes) -> char* {
    char* p = base + off;
    off += (bytes + 255) & ~(size_t)255;
    return p;
  };
  bf16*  WT_in  = (bf16*)carve((size_t)(2 * D_INNER) * D_MODEL * 2);  // [4096,1024]
  bf16*  WT_x   = (bf16*)carve((size_t)XDBL_N * D_INNER * 2);         // [96,2048]
  bf16*  WT_dt  = (bf16*)carve((size_t)D_INNER * DT_RANK * 2);        // [2048,64]
  bf16*  WT_out = (bf16*)carve((size_t)D_MODEL * D_INNER * 2);        // [1024,2048]
  bf16*  xbf    = (bf16*)carve((size_t)M_ROWS * D_MODEL * 2);
  float* y1     = (float*)carve((size_t)M_ROWS * 2 * D_INNER * 4);    // u_raw | res
  float* u      = (float*)carve((size_t)M_ROWS * D_INNER * 4);
  bf16*  ubf    = (bf16*)carve((size_t)M_ROWS * D_INNER * 2);
  float* xdbl   = (float*)carve((size_t)M_ROWS * XDBL_N * 4);
  bf16*  dtabf  = (bf16*)carve((size_t)M_ROWS * DT_RANK * 2);
  float* delta  = (float*)carve((size_t)M_ROWS * D_INNER * 4);
  bf16*  ybf    = (bf16*)carve((size_t)M_ROWS * D_INNER * 2);

  const dim3 blk(256);
  auto blocks = [](long long n) { return dim3((unsigned)((n + 255) / 256)); };

  // 1) weight transposes -> bf16 W^T
  k_transpose_bf16<<<blocks((long long)D_MODEL * 2 * D_INNER), blk, 0, stream>>>(
      W_in, WT_in, D_MODEL, 2 * D_INNER);
  k_transpose_bf16<<<blocks((long long)D_INNER * XDBL_N), blk, 0, stream>>>(
      W_x, WT_x, D_INNER, XDBL_N);
  k_transpose_bf16<<<blocks((long long)DT_RANK * D_INNER), blk, 0, stream>>>(
      W_dt, WT_dt, DT_RANK, D_INNER);
  k_transpose_bf16<<<blocks((long long)D_INNER * D_MODEL), blk, 0, stream>>>(
      W_out, WT_out, D_INNER, D_MODEL);

  // 2) gather-convert x (t,b,c) -> (b*T+t, c) bf16
  k_gather_x_bf16<<<blocks((long long)M_ROWS * D_MODEL), blk, 0, stream>>>(x, xbf);

  // 3) in_proj GEMM (LDS/async pipeline): [8192,1024]x[1024,4096] -> y1
  gemm_bf16_lds<0><<<dim3((M_ROWS / 128) * ((2 * D_INNER) / 64)), blk, 0, stream>>>(
      xbf, WT_in, M_ROWS, 2 * D_INNER, D_MODEL, y1, nullptr);

  // 4) causal depthwise conv + SiLU
  k_conv_silu<<<blocks((long long)M_ROWS * D_INNER), blk, 0, stream>>>(
      y1, conv_w, conv_b, u, ubf);

  // 5) x_proj GEMM (direct, N=96): [8192,2048]x[2048,96] -> x_dbl
  {
    long long tiles = (long long)(M_ROWS / 32) * (XDBL_N / 32);
    gemm_bf16_direct<0><<<dim3((unsigned)((tiles + 7) / 8)), blk, 0, stream>>>(
        ubf, WT_x, M_ROWS, XDBL_N, D_INNER, xdbl, nullptr);
  }

  // 6) dt_proj GEMM + bias + softplus: [8192,64]x[64,2048] -> delta
  k_slice_dta<<<blocks((long long)M_ROWS * DT_RANK), blk, 0, stream>>>(xdbl, dtabf);
  gemm_bf16_lds<1><<<dim3((M_ROWS / 128) * (D_INNER / 64)), blk, 0, stream>>>(
      dtabf, WT_dt, M_ROWS, D_INNER, DT_RANK, delta, b_dt);

  // 7) fused selective scan + skip + gate -> ybf
  k_scan<<<blocks((long long)B_SZ * D_INNER), blk, 0, stream>>>(
      delta, u, xdbl, y1, A_log, Dvec, ybf);

  // 8) out_proj GEMM with (b,t)->(t,b) scatter: [8192,2048]x[2048,1024] -> out
  gemm_bf16_lds<2><<<dim3((M_ROWS / 128) * (D_MODEL / 64)), blk, 0, stream>>>(
      ybf, WT_out, M_ROWS, D_MODEL, D_INNER, out, nullptr);
}
